// TopKEncoderDecoder_53747220742704
// MI455X (gfx1250) — matrix operand order repaired and checked by reading, not compiled, not run on hardware
//
#include <hip/hip_runtime.h>
#include <hip/hip_bf16.h>

typedef __attribute__((ext_vector_type(16))) __bf16   v16bf;
typedef __attribute__((ext_vector_type(8)))  float    v8f;
typedef __attribute__((ext_vector_type(4)))  unsigned uint4v;   // 16B aligned

#define N_ROWS  8192
#define DMODEL  768
#define DSAE    24576
#define TOPK    64

#define BM 128
#define BN 128
#define BK 32
#define LSTRW 20   // LDS row stride in dwords (80B: 16B aligned, bank-friendly)

// RNE f32 -> bf16 bits (values are finite; no NaN care needed)
__device__ __forceinline__ unsigned bf16_bits(float f) {
    unsigned b = __float_as_uint(f);
    return (b + 0x7FFFu + ((b >> 16) & 1u)) >> 16;
}
// split f into bf16 hi + residual lo; return packed {hi0|hi1<<16, lo0|lo1<<16}
__device__ __forceinline__ void split_pack2(float f0, float f1,
                                            unsigned& hp, unsigned& lp) {
    unsigned h0 = bf16_bits(f0), h1 = bf16_bits(f1);
    float r0 = f0 - __uint_as_float(h0 << 16);
    float r1 = f1 - __uint_as_float(h1 << 16);
    unsigned l0 = bf16_bits(r0), l1 = bf16_bits(r1);
    hp = h0 | (h1 << 16);
    lp = l0 | (l1 << 16);
}

union Frag {
    uint4v q[2];
    v16bf  v;
};
// load 16 bf16 halves as two aligned 16B dword-quads from LDS
__device__ __forceinline__ v16bf load_frag(const unsigned* base, int d0, int d1) {
    Frag f;
    f.q[0] = *(const uint4v*)(base + d0);
    f.q[1] = *(const uint4v*)(base + d1);
    return f.v;
}

// ---------------------------------------------------------------------------
// Kernel 1: pre_act = x @ W_enc + b_enc via bf16-split WMMA (~fp32 accurate)
// Block = 256 threads (8 waves). 128x128 tile, K chunks of 32.
// Wave w: wm = w&3 -> 32-row slab (2 m-tiles); wn = w>>2 -> 64-col slab (4 n-tiles).
// 24 v_wmma per K-chunk per wave, 1:1 with ds_load_b128.
// ---------------------------------------------------------------------------
__global__ __launch_bounds__(256)
void sae_encoder_gemm(const float* __restrict__ x,
                      const float* __restrict__ W_enc,
                      const float* __restrict__ b_enc,
                      float* __restrict__ pre_act)
{
    __shared__ unsigned Ahi[BM][LSTRW];
    __shared__ unsigned Alo[BM][LSTRW];
    __shared__ unsigned Bhi[BN][LSTRW];   // transposed: [n][k]
    __shared__ unsigned Blo[BN][LSTRW];

    const int t    = threadIdx.x;
    const int lane = t & 31;
    const int w    = t >> 5;
    const int wm   = w & 3;    // 32-row slab
    const int wn   = w >> 2;   // 64-col slab

    const int m0 = blockIdx.y * BM;
    const int n0 = blockIdx.x * BN;

    v8f acc[2][4];
    #pragma unroll
    for (int mt = 0; mt < 2; ++mt)
        #pragma unroll
        for (int nt = 0; nt < 4; ++nt)
            acc[mt][nt] = (v8f){};

    for (int kb = 0; kb < DMODEL; kb += BK) {
        // --- stage A tile 128x32: thread loads 16 consecutive k floats ---
        {
            const int r  = t >> 1;          // 0..127
            const int kc = (t & 1) * 16;    // 0 or 16
            const float* src = x + (size_t)(m0 + r) * DMODEL + kb + kc;
            #pragma unroll
            for (int p = 0; p < 8; ++p) {
                unsigned hp, lp;
                split_pack2(src[2 * p], src[2 * p + 1], hp, lp);
                Ahi[r][(kc >> 1) + p] = hp;
                Alo[r][(kc >> 1) + p] = lp;
            }
        }
        // --- stage B tile 32x128 -> transposed [n][k]; pack k-pairs ---
        {
            const int q  = t >> 4;          // k-pair 0..15 -> k rows 2q, 2q+1
            const int nc = (t & 15) * 8;    // 0..120
            const float* s0 = W_enc + (size_t)(kb + 2 * q) * DSAE + n0 + nc;
            const float* s1 = s0 + DSAE;
            #pragma unroll
            for (int i = 0; i < 8; ++i) {
                unsigned hp, lp;
                split_pack2(s0[i], s1[i], hp, lp);
                Bhi[nc + i][q] = hp;
                Blo[nc + i][q] = lp;
            }
        }
        __syncthreads();

        // --- A fragments for both m-tiles (contiguous k: k = 16h + s) ---
        const int akw = (lane < 16) ? 0 : 8;   // dword offset within row
        v16bf a_hi[2], a_lo[2];
        #pragma unroll
        for (int mt = 0; mt < 2; ++mt) {
            const int arow = (wm * 32 + mt * 16 + (lane & 15)) * LSTRW;
            a_hi[mt] = load_frag(&Ahi[0][0] + arow, akw, akw + 4);
            a_lo[mt] = load_frag(&Alo[0][0] + arow, akw, akw + 4);
        }
        // --- per n-tile: B fragments + 6 WMMAs (hi*hi + hi*lo + lo*hi) ---
        const int bkw = (lane < 16) ? 0 : 8;   // dword offset within row
        #pragma unroll
        for (int nt = 0; nt < 4; ++nt) {
            const int brow = (wn * 64 + nt * 16 + (lane & 15)) * LSTRW;
            v16bf b_hi = load_frag(&Bhi[0][0] + brow, bkw, bkw + 4);
            v16bf b_lo = load_frag(&Blo[0][0] + brow, bkw, bkw + 4);
            #pragma unroll
            for (int mt = 0; mt < 2; ++mt) {
                acc[mt][nt] = __builtin_amdgcn_wmma_f32_16x16x32_bf16(false, a_hi[mt], false, b_hi, (short)0, acc[mt][nt], false, false);
                acc[mt][nt] = __builtin_amdgcn_wmma_f32_16x16x32_bf16(false, a_hi[mt], false, b_lo, (short)0, acc[mt][nt], false, false);
                acc[mt][nt] = __builtin_amdgcn_wmma_f32_16x16x32_bf16(false, a_lo[mt], false, b_hi, (short)0, acc[mt][nt], false, false);
            }
        }
        __syncthreads();
    }

    // --- store C + bias (f32 C/D layout: lane<16 -> M=r, else M=r+8) ---
    #pragma unroll
    for (int nt = 0; nt < 4; ++nt) {
        const int col = n0 + wn * 64 + nt * 16 + (lane & 15);
        const float bias = b_enc[col];
        #pragma unroll
        for (int mt = 0; mt < 2; ++mt) {
            const int rbase = m0 + wm * 32 + mt * 16 + ((lane < 16) ? 0 : 8);
            #pragma unroll
            for (int r = 0; r < 8; ++r)
                pre_act[(size_t)(rbase + r) * DSAE + col] = acc[mt][nt][r] + bias;
        }
    }
}

// ---------------------------------------------------------------------------
// Kernel 2: per-row exact top-64 (radix select on order-preserving uint keys,
// whole 96KB row in LDS), sparsify latents in place, fused sparse decode.
// One block (256 threads) per row.
// ---------------------------------------------------------------------------
__device__ __forceinline__ unsigned mapf(float f) {
    unsigned b = __float_as_uint(f);
    return (b & 0x80000000u) ? ~b : (b | 0x80000000u);
}
__device__ __forceinline__ float unmapf(unsigned u) {
    unsigned b = (u & 0x80000000u) ? (u ^ 0x80000000u) : ~u;
    return __uint_as_float(b);
}

__global__ __launch_bounds__(256)
void sae_topk_decode(float* __restrict__ latents,      // in: pre_act, out: sparse
                     const float* __restrict__ W_dec,
                     float* __restrict__ out)
{
    extern __shared__ unsigned smem[];
    unsigned* keys = smem;            // DSAE keys
    unsigned* hist = smem + DSAE;     // 256 bins

    __shared__ unsigned s_prefix, s_remk;
    __shared__ int s_eqCnt, s_listCnt;
    __shared__ int   lidx[TOPK + 64];
    __shared__ float lval[TOPK + 64];

    const int row = blockIdx.x;
    const int t   = threadIdx.x;
    float* lat = latents + (size_t)row * DSAE;

    for (int i = t; i < DSAE; i += 256)
        keys[i] = mapf(lat[i]);

    if (t == 0) { s_prefix = 0u; s_remk = TOPK; s_eqCnt = 0; s_listCnt = 0; }
    __syncthreads();

    // 4-pass MSB-first radix select: exact 32-bit threshold of the 64th value
    for (int pass = 3; pass >= 0; --pass) {
        hist[t] = 0;
        __syncthreads();
        const unsigned prefix = s_prefix;
        const unsigned pmask  = (pass == 3) ? 0u : (0xFFFFFFFFu << ((pass + 1) * 8));
        for (int i = t; i < DSAE; i += 256) {
            unsigned u = keys[i];
            if ((u & pmask) == prefix)
                atomicAdd(&hist[(u >> (pass * 8)) & 255u], 1u);
        }
        __syncthreads();
        if (t == 0) {
            unsigned remk = s_remk, cum = 0;
            int b = 255;
            for (; b > 0; --b) {
                if (cum + hist[b] >= remk) break;
                cum += hist[b];
            }
            s_prefix = prefix | ((unsigned)b << (pass * 8));
            s_remk   = remk - cum;   // still needed within this bin
        }
        __syncthreads();
    }

    const unsigned thr  = s_prefix;
    const int      need = (int)s_remk;   // # elements equal to threshold to keep
    __syncthreads();

    // sparsify: keep u>thr plus `need` ties; rewrite full row; collect (idx,val)
    for (int i = t; i < DSAE; i += 256) {
        const unsigned u = keys[i];
        bool keep = (u > thr);
        if (!keep && u == thr)
            keep = (atomicAdd(&s_eqCnt, 1) < need);
        const float f = unmapf(u);
        if (keep) {
            int p = atomicAdd(&s_listCnt, 1);
            if (p < TOPK + 64) { lidx[p] = i; lval[p] = f; }
            lat[i] = f;
        } else {
            lat[i] = 0.0f;
        }
    }
    __syncthreads();

    // fused sparse decode: out[row,:] = sum_j lval[j] * W_dec[lidx[j], :]
    const int cnt = min(s_listCnt, TOPK + 64);
    float a0 = 0.f, a1 = 0.f, a2 = 0.f;
    for (int j = 0; j < cnt; ++j) {
        const float v = lval[j];
        const float* wd = W_dec + (size_t)lidx[j] * DMODEL;
        a0 += v * wd[t];
        a1 += v * wd[t + 256];
        a2 += v * wd[t + 512];
    }
    float* o = out + (size_t)row * DMODEL;
    o[t]       = a0;
    o[t + 256] = a1;
    o[t + 512] = a2;
}

// ---------------------------------------------------------------------------
extern "C" void kernel_launch(void* const* d_in, const int* in_sizes, int n_in,
                              void* d_out, int out_size, void* d_ws, size_t ws_size,
                              hipStream_t stream)
{
    const float* x     = (const float*)d_in[0];
    const float* W_enc = (const float*)d_in[1];
    const float* b_enc = (const float*)d_in[2];
    const float* W_dec = (const float*)d_in[3];

    float* out     = (float*)d_out;                       // [8192, 768]
    float* latents = out + (size_t)N_ROWS * DMODEL;       // [8192, 24576]

    dim3 g1(DSAE / BN, N_ROWS / BM);                      // 192 x 64
    sae_encoder_gemm<<<g1, 256, 0, stream>>>(x, W_enc, b_enc, latents);

    const size_t lds2 = (size_t)(DSAE + 256) * sizeof(unsigned);  // ~97 KB of 320 KB
    sae_topk_decode<<<N_ROWS, 256, lds2, stream>>>(latents, W_dec, out);
}
